// EHRMamba2_39187281609377
// MI455X (gfx1250) — matrix operand order repaired and checked by reading, compiled, not run on hardware
//
#include <hip/hip_runtime.h>

// ---------------- model constants ----------------
#define B_      2
#define L_      2048
#define HID_    768
#define NH_     24
#define PD_     64
#define NS_     64
#define KCONV_  4
#define NL_     8
#define INTER_  1536
#define CONV_   1664        // INTER + 2*NS
#define DIN_    3224        // 2*INTER + 2*NS + NH
#define ROWS_   (B_ * L_)   // 4096
#define EPS_    1e-5f

// ---------------- bf16 / vector types ----------------
typedef __bf16 bf16;
typedef bf16  bf16v8  __attribute__((ext_vector_type(8)));
typedef bf16  bf16v16 __attribute__((ext_vector_type(16)));
typedef float fv8     __attribute__((ext_vector_type(8)));

__device__ __forceinline__ bf16 f2bf(float f) {
    union { float f; unsigned u; } a; a.f = f;
    unsigned u = a.u + 0x7FFFu + ((a.u >> 16) & 1u);   // RNE
    union { unsigned short s; bf16 b; } r;
    r.s = (unsigned short)(u >> 16);
    return r.b;
}

// ---------------- fp32 -> bf16 convert ----------------
__global__ __launch_bounds__(256) void f2bf_kernel(const float* __restrict__ src,
                                                   bf16* __restrict__ dst, int n) {
    int i = blockIdx.x * 256 + threadIdx.x;
    if (i < n) dst[i] = f2bf(src[i]);
}

// ---------------- embedding gather ----------------
__global__ __launch_bounds__(256) void embed_kernel(const int* __restrict__ ids,
                                                    const float* __restrict__ emb,
                                                    float* __restrict__ h) {
    int idx = blockIdx.x * 256 + threadIdx.x;
    if (idx >= ROWS_ * HID_) return;
    int row = idx / HID_, c = idx - row * HID_;
    h[idx] = emb[(size_t)ids[row] * HID_ + c];
}

// ---------------- rmsnorm -> bf16 ----------------
__global__ __launch_bounds__(256) void rmsnorm_bf16_kernel(const float* __restrict__ x,
                                                           const float* __restrict__ w,
                                                           bf16* __restrict__ out) {
    __shared__ float red[256];
    int row = blockIdx.x, tid = threadIdx.x;
    const float* xr = x + (size_t)row * HID_;
    float ss = 0.f;
    for (int c = tid; c < HID_; c += 256) { float v = xr[c]; ss += v * v; }
    red[tid] = ss; __syncthreads();
    for (int s = 128; s > 0; s >>= 1) { if (tid < s) red[tid] += red[tid + s]; __syncthreads(); }
    float rs = rsqrtf(red[0] / (float)HID_ + EPS_);
    for (int c = tid; c < HID_; c += 256) out[(size_t)row * HID_ + c] = f2bf(xr[c] * rs * w[c]);
}

// ---------------- WMMA bf16 GEMM:  C(M x N) = A(M x K) * W(N x K)^T  (+C) ----
// A operand (16x32, 16-bit) per ISA: lane m=lane&15, half=lane>>4:
//   a[0..7]  = A[m][kk + half*8 + 0..7]
//   a[8..15] = A[m][kk + half*8 + 16 .. +23]        (two contiguous 8-elt runs)
// B operand (32x16): lane n=lane&15:
//   b[0..7]  = W[n][kk + half*8 + 0..7]
//   b[8..15] = W[n][kk + half*8 + 16 .. +23]        (same two 8-elt runs as A)
__device__ __forceinline__ bf16v16 loadA(const bf16* __restrict__ row, int kk, int half) {
    const bf16v8* p = (const bf16v8*)(row + kk + half * 8);
    bf16v8 lo = p[0];
    bf16v8 hi = p[2];   // +16 elements
    return __builtin_shufflevector(lo, hi, 0,1,2,3,4,5,6,7,8,9,10,11,12,13,14,15);
}

__device__ __forceinline__ void store_tile(const fv8& acc, float* __restrict__ C, int ldc,
                                           int mbase, int nbase, int lane, int Ntot, int accum) {
    int col = nbase + (lane & 15);
    if (col >= Ntot) return;
    float* cp = C + (size_t)(mbase + (lane >> 4) * 8) * ldc + col;
#pragma unroll
    for (int r = 0; r < 8; ++r) {
        float v = acc[r];
        if (accum) v += cp[(size_t)r * ldc];
        cp[(size_t)r * ldc] = v;
    }
}

__global__ __launch_bounds__(256) void gemm_bf16_kernel(const bf16* __restrict__ A,
                                                        const bf16* __restrict__ W,
                                                        float* __restrict__ C,
                                                        int Kdim, int Ntot, int ldc,
                                                        int mt2, int nt2, int accum) {
    int wave = blockIdx.x * 8 + (threadIdx.x >> 5);
    int lane = threadIdx.x & 31;
    int tm2 = wave % mt2;
    int tn2 = wave / mt2;
    if (tn2 >= nt2) return;
    int m0 = tm2 * 32, n0 = tn2 * 32;
    int m15 = lane & 15, half = lane >> 4;

    const bf16* arow0 = A + (size_t)(m0 + m15) * Kdim;
    const bf16* arow1 = arow0 + (size_t)16 * Kdim;
    int nA = n0 + m15, nB = n0 + 16 + m15;
    const bf16* wrow0 = W + (size_t)(nA < Ntot ? nA : Ntot - 1) * Kdim;  // clamped; bad cols masked at store
    const bf16* wrow1 = W + (size_t)(nB < Ntot ? nB : Ntot - 1) * Kdim;

    fv8 acc00 = {}, acc01 = {}, acc10 = {}, acc11 = {};
    for (int kk = 0; kk < Kdim; kk += 32) {
        if (kk + 32 < Kdim) {
            __builtin_prefetch(arow0 + kk + 32, 0, 3);
            __builtin_prefetch(wrow0 + kk + 32, 0, 3);
        }
        bf16v16 a0 = loadA(arow0, kk, half);
        bf16v16 a1 = loadA(arow1, kk, half);
        bf16v16 b0 = loadA(wrow0, kk, half);
        bf16v16 b1 = loadA(wrow1, kk, half);
        acc00 = __builtin_amdgcn_wmma_f32_16x16x32_bf16(false, a0, false, b0, (short)0, acc00, false, false);
        acc01 = __builtin_amdgcn_wmma_f32_16x16x32_bf16(false, a0, false, b1, (short)0, acc01, false, false);
        acc10 = __builtin_amdgcn_wmma_f32_16x16x32_bf16(false, a1, false, b0, (short)0, acc10, false, false);
        acc11 = __builtin_amdgcn_wmma_f32_16x16x32_bf16(false, a1, false, b1, (short)0, acc11, false, false);
    }
    store_tile(acc00, C, ldc, m0,      n0,      lane, Ntot, accum);
    store_tile(acc01, C, ldc, m0,      n0 + 16, lane, Ntot, accum);
    store_tile(acc10, C, ldc, m0 + 16, n0,      lane, Ntot, accum);
    store_tile(acc11, C, ldc, m0 + 16, n0 + 16, lane, Ntot, accum);
}

// ---------------- causal depthwise conv (K=4) + bias + SiLU ----------------
__global__ __launch_bounds__(256) void conv_silu_kernel(const float* __restrict__ proj,
                                                        const float* __restrict__ cw,
                                                        const float* __restrict__ cb,
                                                        float* __restrict__ hbc) {
    int idx = blockIdx.x * 256 + threadIdx.x;
    if (idx >= ROWS_ * CONV_) return;
    int row = idx / CONV_, c = idx - row * CONV_;
    int l = row & (L_ - 1);
    float acc = cb[c];
#pragma unroll
    for (int k = 0; k < KCONV_; ++k) {
        int ls = l - (KCONV_ - 1) + k;
        if (ls >= 0)
            acc += cw[c * KCONV_ + k] * proj[(size_t)(row - (KCONV_ - 1) + k) * DIN_ + INTER_ + c];
    }
    hbc[idx] = acc / (1.f + __expf(-acc));
}

// ---------------- selective scan: one block per (b, head) ----------------
__global__ __launch_bounds__(256) void scan_kernel(const float* __restrict__ proj,
                                                   const float* __restrict__ hbc,
                                                   const float* __restrict__ dt_bias,
                                                   const float* __restrict__ A_log,
                                                   const float* __restrict__ Dp,
                                                   float* __restrict__ y) {
    int hidx = blockIdx.x % NH_;
    int b    = blockIdx.x / NH_;
    float Ac  = -__expf(A_log[hidx]);
    float Dc  = Dp[hidx];
    float dtb = dt_bias[hidx];
    __shared__ float xs[PD_], Bs[NS_], Cs[NS_];
    __shared__ float sdt;
    int tid = threadIdx.x;
    int p = tid >> 2, nb = (tid & 3) * 16;
    float s[16];
#pragma unroll
    for (int j = 0; j < 16; ++j) s[j] = 0.f;

    for (int t = 0; t < L_; ++t) {
        int row = b * L_ + t;
        const float* hr = hbc + (size_t)row * CONV_;
        if (tid < 64)          xs[tid]        = hr[hidx * PD_ + tid];
        else if (tid < 128)    Bs[tid - 64]   = hr[INTER_ + (tid - 64)];
        else if (tid < 192)    Cs[tid - 128]  = hr[INTER_ + NS_ + (tid - 128)];
        else if (tid == 192) {
            float d = proj[(size_t)row * DIN_ + (INTER_ + CONV_) + hidx] + dtb;
            sdt = (d > 20.f) ? d : log1pf(__expf(d));   // softplus
        }
        __syncthreads();
        float dtv = sdt;
        float da  = __expf(dtv * Ac);
        float xv  = xs[p];
        float dtx = dtv * xv;
        float partial = 0.f;
#pragma unroll
        for (int j = 0; j < 16; ++j) {
            s[j] = s[j] * da + dtx * Bs[nb + j];
            partial += s[j] * Cs[nb + j];
        }
        partial += __shfl_xor(partial, 1);
        partial += __shfl_xor(partial, 2);
        if ((tid & 3) == 0)
            y[(size_t)row * INTER_ + hidx * PD_ + p] = partial + Dc * xv;
        __syncthreads();
    }
}

// ---------------- gate * SiLU, rmsnorm(gn_w) -> bf16 ----------------
__global__ __launch_bounds__(256) void gatenorm_kernel(const float* __restrict__ y,
                                                       const float* __restrict__ proj,
                                                       const float* __restrict__ gnw,
                                                       bf16* __restrict__ out) {
    __shared__ float tmp[INTER_];
    __shared__ float red[256];
    int row = blockIdx.x, tid = threadIdx.x;
    const float* yr = y + (size_t)row * INTER_;
    const float* pr = proj + (size_t)row * DIN_;
    float ss = 0.f;
    for (int c = tid; c < INTER_; c += 256) {
        float g = pr[c];
        float v = yr[c] * (g / (1.f + __expf(-g)));
        tmp[c] = v; ss += v * v;
    }
    red[tid] = ss; __syncthreads();
    for (int s2 = 128; s2 > 0; s2 >>= 1) { if (tid < s2) red[tid] += red[tid + s2]; __syncthreads(); }
    float rs = rsqrtf(red[0] / (float)INTER_ + EPS_);
    for (int c = tid; c < INTER_; c += 256)
        out[(size_t)row * INTER_ + c] = f2bf(tmp[c] * rs * gnw[c]);
}

// ---------------- final rmsnorm(cls) + dense/gelu + head ----------------
__global__ __launch_bounds__(256) void head_kernel(const float* __restrict__ h,
                                                   const float* __restrict__ normfw,
                                                   const float* __restrict__ dw,
                                                   const float* __restrict__ db,
                                                   const float* __restrict__ hw,
                                                   const float* __restrict__ hb,
                                                   float* __restrict__ out) {
    __shared__ float cls[HID_], z[HID_], red[256];
    int tid = threadIdx.x;
    for (int b = 0; b < B_; ++b) {
        const float* hr = h + (size_t)(b * L_) * HID_;
        float ss = 0.f;
        for (int c = tid; c < HID_; c += 256) { float v = hr[c]; ss += v * v; }
        red[tid] = ss; __syncthreads();
        for (int s2 = 128; s2 > 0; s2 >>= 1) { if (tid < s2) red[tid] += red[tid + s2]; __syncthreads(); }
        float rs = rsqrtf(red[0] / (float)HID_ + EPS_);
        for (int c = tid; c < HID_; c += 256) cls[c] = hr[c] * rs * normfw[c];
        __syncthreads();
        for (int j = tid; j < HID_; j += 256) {
            float a = db[j];
            const float* wr = dw + (size_t)j * HID_;
            for (int k = 0; k < HID_; ++k) a += cls[k] * wr[k];
            z[j] = 0.5f * a * (1.f + erff(a * 0.70710678118f));   // exact gelu
        }
        __syncthreads();
        if (tid < 2) {
            float a = hb[tid];
            const float* wr = hw + (size_t)tid * HID_;
            for (int k = 0; k < HID_; ++k) a += z[k] * wr[k];
            out[b * 2 + tid] = a;
        }
        __syncthreads();
    }
}

// ---------------- host launcher ----------------
extern "C" void kernel_launch(void* const* d_in, const int* in_sizes, int n_in,
                              void* d_out, int out_size, void* d_ws, size_t ws_size,
                              hipStream_t stream) {
    (void)in_sizes; (void)n_in; (void)out_size; (void)ws_size;
    const int*   ids       = (const int*)  d_in[0];
    const float* emb       = (const float*)d_in[1];
    const float* norm_w    = (const float*)d_in[2];
    const float* in_proj_w = (const float*)d_in[3];
    const float* conv_w    = (const float*)d_in[4];
    const float* conv_b    = (const float*)d_in[5];
    const float* dt_bias   = (const float*)d_in[6];
    const float* A_log     = (const float*)d_in[7];
    const float* D_param   = (const float*)d_in[8];
    const float* gn_w      = (const float*)d_in[9];
    const float* out_w     = (const float*)d_in[10];
    const float* normf_w   = (const float*)d_in[11];
    const float* dense_w   = (const float*)d_in[12];
    const float* dense_b   = (const float*)d_in[13];
    const float* head_w    = (const float*)d_in[14];
    const float* head_b    = (const float*)d_in[15];

    char* ws = (char*)d_ws;
    size_t off = 0;
    auto take = [&](size_t bytes) { size_t o = off; off = (off + bytes + 255) & ~(size_t)255; return o; };
    const int nWin  = NL_ * DIN_ * HID_;     // 19,808,256
    const int nWout = NL_ * HID_ * INTER_;   //  9,437,184
    bf16*  Win    = (bf16*)(ws + take((size_t)nWin * 2));
    bf16*  Wout   = (bf16*)(ws + take((size_t)nWout * 2));
    float* h      = (float*)(ws + take((size_t)ROWS_ * HID_ * 4));
    bf16*  normbf = (bf16*)(ws + take((size_t)ROWS_ * HID_ * 2));
    float* proj   = (float*)(ws + take((size_t)ROWS_ * DIN_ * 4));
    float* hbc    = (float*)(ws + take((size_t)ROWS_ * CONV_ * 4));
    float* ybuf   = (float*)(ws + take((size_t)ROWS_ * INTER_ * 4));
    bf16*  ygb    = (bf16*)(ws + take((size_t)ROWS_ * INTER_ * 2));

    // one-time (per call) weight conversion to bf16
    f2bf_kernel<<<(nWin + 255) / 256, 256, 0, stream>>>(in_proj_w, Win, nWin);
    f2bf_kernel<<<(nWout + 255) / 256, 256, 0, stream>>>(out_w, Wout, nWout);

    embed_kernel<<<(ROWS_ * HID_ + 255) / 256, 256, 0, stream>>>(ids, emb, h);

    const int mt2 = ROWS_ / 32;   // 128
    for (int i = 0; i < NL_; ++i) {
        rmsnorm_bf16_kernel<<<ROWS_, 256, 0, stream>>>(h, norm_w + i * HID_, normbf);

        int nt2a = (DIN_ + 31) / 32;  // 101
        int blocksA = (mt2 * nt2a + 7) / 8;
        gemm_bf16_kernel<<<blocksA, 256, 0, stream>>>(normbf, Win + (size_t)i * DIN_ * HID_,
                                                      proj, HID_, DIN_, DIN_, mt2, nt2a, 0);

        conv_silu_kernel<<<(ROWS_ * CONV_ + 255) / 256, 256, 0, stream>>>(
            proj, conv_w + i * CONV_ * KCONV_, conv_b + i * CONV_, hbc);

        scan_kernel<<<B_ * NH_, 256, 0, stream>>>(proj, hbc, dt_bias + i * NH_,
                                                  A_log + i * NH_, D_param + i * NH_, ybuf);

        gatenorm_kernel<<<ROWS_, 256, 0, stream>>>(ybuf, proj, gn_w + i * INTER_, ygb);

        int nt2b = HID_ / 32;         // 24
        int blocksB = (mt2 * nt2b + 7) / 8;
        gemm_bf16_kernel<<<blocksB, 256, 0, stream>>>(ygb, Wout + (size_t)i * HID_ * INTER_,
                                                      h, INTER_, HID_, HID_, mt2, nt2b, 1);
    }

    head_kernel<<<1, 256, 0, stream>>>(h, normf_w, dense_w, dense_b, head_w, head_b, (float*)d_out);
}
